// TalkingMultiheadSelfAttention_8796093022421
// MI455X (gfx1250) — compile-verified, matrix operand order checked by reading, not run on hardware
//
#include <hip/hip_runtime.h>
#include <hip/hip_bf16.h>

// ---------------------------------------------------------------------------
// Talking-heads multi-head self-attention for MI455X (gfx1250, wave32, WMMA)
//   B=32, N=512, DIM=512, HEADS=8, hd=64
// ---------------------------------------------------------------------------

#define BQ    32
#define NSEQ  512
#define DIMC  512
#define NHEAD 8
#define HD    64

typedef __attribute__((ext_vector_type(16))) __bf16 v16bf;
typedef __attribute__((ext_vector_type(8)))  __bf16 v8bf;
typedef __attribute__((ext_vector_type(8)))  float  v8f;

__device__ __forceinline__ v8f wmma_bf16(v16bf a, v16bf b, v8f c) {
  // D = A(16x32 bf16) x B(32x16 bf16) + C(16x16 f32)
  return __builtin_amdgcn_wmma_f32_16x16x32_bf16(false, a, false, b, (short)0, c,
                                                 false, false);
}

// Build a 16x32 (K-slice) A/B fragment for one lane.
// Layout (wave32, 16-bit): lane L holds row/col L%16; with hlf = L/16:
//   elements e=0..7  -> K = kbase + 8*hlf + e        (contiguous 16B)
//   elements e=8..15 -> K = kbase + 16 + 8*hlf + (e-8) (contiguous 16B)
__device__ __forceinline__ v16bf ld_frag(const __bf16* rowbase, int hlf, int kbase) {
  union { v16bf v; v8bf h[2]; } f;
  f.h[0] = *(const v8bf*)(rowbase + kbase + 8 * hlf);
  f.h[1] = *(const v8bf*)(rowbase + kbase + 16 + 8 * hlf);
  return f.v;
}

__device__ __forceinline__ v8f zero_v8f() {
  v8f z = {0.f, 0.f, 0.f, 0.f, 0.f, 0.f, 0.f, 0.f};
  return z;
}

// ---------------------------------------------------------------------------
// Generic fp32->bf16 WMMA GEMM:  C(M x NOUT) = A(M x 512) * W(512 x NOUT)
// Block tile 128x64, 256 threads = 8 waves in 4(M) x 2(N); wave tile 32x32.
// MODE 0: qkv epilogue (split/scale into q,k bf16 (B,H,N,64) and vT (B,H,64,N))
// MODE 1: plain epilogue out[row*512+col] = acc + bias[col]
// ---------------------------------------------------------------------------
template <int NOUT, int MODE>
__global__ __launch_bounds__(256) void gemm_bf16_kernel(
    const float* __restrict__ A, const float* __restrict__ W,
    const float* __restrict__ bias,
    __bf16* __restrict__ qws, __bf16* __restrict__ kws, __bf16* __restrict__ vtws,
    float* __restrict__ outp) {
  __shared__ __bf16 lds_a[128][32];   // A tile, row-major (M x K)
  __shared__ __bf16 lds_bt[64][32];   // W tile, transposed (N x K)

  const int tid  = threadIdx.x;
  const int lane = tid & 31;
  const int wv   = tid >> 5;
  const int wm   = wv & 3;      // M quadrant (32 rows)
  const int wn   = wv >> 2;     // N half (32 cols)
  const int hlf  = lane >> 4;
  const int l16  = lane & 15;
  const int bm   = blockIdx.y * 128;
  const int bn   = blockIdx.x * 64;

  v8f acc[2][2];
#pragma unroll
  for (int i = 0; i < 2; ++i)
#pragma unroll
    for (int j = 0; j < 2; ++j) acc[i][j] = zero_v8f();

  for (int k0 = 0; k0 < DIMC; k0 += 32) {
    // ---- stage A tile: each thread converts 16 contiguous fp32 -> bf16
    {
      const int ar = tid >> 1;
      const int ac = (tid & 1) * 16;
      const float* src = A + (size_t)(bm + ar) * DIMC + k0 + ac;
#pragma unroll
      for (int jj = 0; jj < 4; ++jj) {
        float4 t = ((const float4*)src)[jj];
        lds_a[ar][ac + 4 * jj + 0] = (__bf16)t.x;
        lds_a[ar][ac + 4 * jj + 1] = (__bf16)t.y;
        lds_a[ar][ac + 4 * jj + 2] = (__bf16)t.z;
        lds_a[ar][ac + 4 * jj + 3] = (__bf16)t.w;
      }
    }
    // ---- stage W tile transposed: thread loads 8 contiguous cols at one k
    {
      const int kb = tid & 31;
      const int nb = (tid >> 5) * 8;
      const float* src = W + (size_t)(k0 + kb) * NOUT + bn + nb;
      float4 t0 = ((const float4*)src)[0];
      float4 t1 = ((const float4*)src)[1];
      lds_bt[nb + 0][kb] = (__bf16)t0.x;
      lds_bt[nb + 1][kb] = (__bf16)t0.y;
      lds_bt[nb + 2][kb] = (__bf16)t0.z;
      lds_bt[nb + 3][kb] = (__bf16)t0.w;
      lds_bt[nb + 4][kb] = (__bf16)t1.x;
      lds_bt[nb + 5][kb] = (__bf16)t1.y;
      lds_bt[nb + 6][kb] = (__bf16)t1.z;
      lds_bt[nb + 7][kb] = (__bf16)t1.w;
    }
    if (k0 + 32 < DIMC) {
      __builtin_prefetch(A + (size_t)(bm + (tid >> 1)) * DIMC + k0 + 32, 0, 1);
      __builtin_prefetch(W + (size_t)(k0 + 32 + (tid & 31)) * NOUT + bn, 0, 1);
    }
    __syncthreads();

    v16bf afr[2], bfr[2];
#pragma unroll
    for (int i = 0; i < 2; ++i)
      afr[i] = ld_frag(&lds_a[wm * 32 + i * 16 + l16][0], hlf, 0);
#pragma unroll
    for (int j = 0; j < 2; ++j)
      bfr[j] = ld_frag(&lds_bt[wn * 32 + j * 16 + l16][0], hlf, 0);
#pragma unroll
    for (int i = 0; i < 2; ++i)
#pragma unroll
      for (int j = 0; j < 2; ++j) acc[i][j] = wmma_bf16(afr[i], bfr[j], acc[i][j]);
    __syncthreads();
  }

  // ---- epilogue. C layout: lane col = l16, VGPR r -> row r + 8*hlf
#pragma unroll
  for (int i = 0; i < 2; ++i) {
#pragma unroll
    for (int j = 0; j < 2; ++j) {
      const int col = bn + wn * 32 + j * 16 + l16;
      const float bv = bias[col];
#pragma unroll
      for (int r = 0; r < 8; ++r) {
        const int row = bm + wm * 32 + i * 16 + r + 8 * hlf;
        const float val = acc[i][j][r] + bv;
        if (MODE == 0) {
          const int b = row >> 9, n = row & 511;          // row = b*512 + n
          const int s = col >> 9, rem = col & 511;        // col = s*512 + h*64 + d
          const int h = rem >> 6, d = rem & 63;
          if (s == 0)
            qws[((size_t)(b * NHEAD + h) * NSEQ + n) * HD + d] = (__bf16)(val * 0.125f);
          else if (s == 1)
            kws[((size_t)(b * NHEAD + h) * NSEQ + n) * HD + d] = (__bf16)val;
          else
            vtws[((size_t)(b * NHEAD + h) * HD + d) * NSEQ + n] = (__bf16)val;
        } else {
          outp[(size_t)row * DIMC + col] = val;
        }
      }
    }
  }
}

// ---------------------------------------------------------------------------
// Pass 1: per (b, 16-row n-tile): wave h computes QK^T (its head) chunk by
// chunk; threads apply talking-heads mix1 + interaction + mask; online
// max/sum-exp over the m axis -> rowmax/rowsum (per post-mix head g, row n).
// ---------------------------------------------------------------------------
__global__ __launch_bounds__(256) void attn_stats_kernel(
    const __bf16* __restrict__ q, const __bf16* __restrict__ kmat,
    const float* __restrict__ interaction, const unsigned char* __restrict__ mask,
    const float* __restrict__ Wt1, const float* __restrict__ bt1,
    float* __restrict__ rowmax, float* __restrict__ rowsum) {
  __shared__ float s_lds[NHEAD][16][64];
  __shared__ float run_max[NHEAD][16];
  __shared__ float run_sum[NHEAD][16];
  __shared__ float wt1[NHEAD][NHEAD];
  __shared__ float bt1s[NHEAD];

  const int b  = blockIdx.y;
  const int n0 = blockIdx.x * 16;
  const int tid  = threadIdx.x;
  const int lane = tid & 31;
  const int h    = tid >> 5;    // wave id == head for score WMMA
  const int hlf  = lane >> 4;
  const int l16  = lane & 15;
  const int tn = tid >> 4;          // 0..15: n within tile
  const int tm = (tid & 15) * 4;    // base m within 64-chunk

  if (tid < 64) wt1[tid >> 3][tid & 7] = Wt1[tid];
  if (tid < NHEAD) bt1s[tid] = bt1[tid];
  if (tid < 128) { run_max[tid >> 4][tid & 15] = -3.0e38f; run_sum[tid >> 4][tid & 15] = 0.f; }

  const __bf16* qrow = q + ((size_t)(b * NHEAD + h) * NSEQ + n0 + l16) * HD;
  const v16bf qf0 = ld_frag(qrow, hlf, 0);
  const v16bf qf1 = ld_frag(qrow, hlf, 32);
  __syncthreads();

  for (int m0 = 0; m0 < NSEQ; m0 += 64) {
    // -- scores for head h, 16x64 chunk (K = 64 -> two WMMA k-steps)
    v8f sacc[4];
#pragma unroll
    for (int j = 0; j < 4; ++j) sacc[j] = zero_v8f();
#pragma unroll
    for (int j = 0; j < 4; ++j) {
      const __bf16* krow =
          kmat + ((size_t)(b * NHEAD + h) * NSEQ + m0 + j * 16 + l16) * HD;
      sacc[j] = wmma_bf16(qf0, ld_frag(krow, hlf, 0), sacc[j]);
      sacc[j] = wmma_bf16(qf1, ld_frag(krow, hlf, 32), sacc[j]);
    }
#pragma unroll
    for (int j = 0; j < 4; ++j)
#pragma unroll
      for (int r = 0; r < 8; ++r)
        s_lds[h][r + 8 * hlf][j * 16 + l16] = sacc[j][r];
    __syncthreads();

    // -- mix1 + interaction + mask -> logits (8 heads x 4 m per thread)
    float lg[NHEAD][4];
#pragma unroll
    for (int mi = 0; mi < 4; ++mi) {
      const int m = m0 + tm + mi;
      float sh[NHEAD];
#pragma unroll
      for (int hh = 0; hh < NHEAD; ++hh) sh[hh] = s_lds[hh][tn][tm + mi];
      const float* ip =
          interaction + (((size_t)(b * NSEQ + n0 + tn)) * NSEQ + m) * NHEAD;
      const float4 i0 = ((const float4*)ip)[0];
      const float4 i1 = ((const float4*)ip)[1];
      const float iv[NHEAD] = {i0.x, i0.y, i0.z, i0.w, i1.x, i1.y, i1.z, i1.w};
      const unsigned char mk = mask[((size_t)b * NSEQ + n0 + tn) * NSEQ + m];
#pragma unroll
      for (int g = 0; g < NHEAD; ++g) {
        float v = bt1s[g];
#pragma unroll
        for (int hh = 0; hh < NHEAD; ++hh) v += sh[hh] * wt1[hh][g];
        v += iv[g];
        lg[g][mi] = mk ? v : -1.0e9f;
      }
    }
    // -- online softmax stats: reduce over the 16 lanes sharing row tn
#pragma unroll
    for (int g = 0; g < NHEAD; ++g) {
      float cm = lg[g][0];
#pragma unroll
      for (int mi = 1; mi < 4; ++mi) cm = fmaxf(cm, lg[g][mi]);
      for (int off = 1; off < 16; off <<= 1) cm = fmaxf(cm, __shfl_xor(cm, off, 32));
      float cs = 0.f;
#pragma unroll
      for (int mi = 0; mi < 4; ++mi) cs += __expf(lg[g][mi] - cm);
      for (int off = 1; off < 16; off <<= 1) cs += __shfl_xor(cs, off, 32);
      if ((tid & 15) == 0) {
        const float rm = run_max[g][tn];
        const float nm = fmaxf(rm, cm);
        run_sum[g][tn] = run_sum[g][tn] * __expf(rm - nm) + cs * __expf(cm - nm);
        run_max[g][tn] = nm;
      }
    }
    __syncthreads();
  }

  if (tid < 128) {
    const int g = tid >> 4, n = tid & 15;
    rowmax[(size_t)(b * NHEAD + g) * NSEQ + n0 + n] = run_max[g][n];
    rowsum[(size_t)(b * NHEAD + g) * NSEQ + n0 + n] = run_sum[g][n];
  }
}

// ---------------------------------------------------------------------------
// Pass 2: recompute scores chunk, softmax with precomputed stats, write
// pure_attention, mix2, then attn @ V via WMMA -> attended (B,N,DIM).
// ---------------------------------------------------------------------------
__global__ __launch_bounds__(256) void attn_apply_kernel(
    const __bf16* __restrict__ q, const __bf16* __restrict__ kmat,
    const __bf16* __restrict__ vt,
    const float* __restrict__ interaction, const unsigned char* __restrict__ mask,
    const float* __restrict__ Wt1, const float* __restrict__ bt1,
    const float* __restrict__ Wt2, const float* __restrict__ bt2,
    const float* __restrict__ rowmax, const float* __restrict__ rowsum,
    float* __restrict__ pure_attn, float* __restrict__ attended) {
  __shared__ float  s_lds[NHEAD][16][64];
  __shared__ __bf16 a_lds[NHEAD][16][64];
  __shared__ float wt1[NHEAD][NHEAD], wt2[NHEAD][NHEAD];
  __shared__ float bt1s[NHEAD], bt2s[NHEAD];

  const int b  = blockIdx.y;
  const int n0 = blockIdx.x * 16;
  const int tid  = threadIdx.x;
  const int lane = tid & 31;
  const int h    = tid >> 5;
  const int hlf  = lane >> 4;
  const int l16  = lane & 15;
  const int tn = tid >> 4;
  const int tm = (tid & 15) * 4;

  if (tid < 64) { wt1[tid >> 3][tid & 7] = Wt1[tid]; wt2[tid >> 3][tid & 7] = Wt2[tid]; }
  if (tid < NHEAD) { bt1s[tid] = bt1[tid]; bt2s[tid] = bt2[tid]; }

  // per-thread softmax stats for its row (all 8 post-mix heads)
  float rm[NHEAD], rsi[NHEAD];
#pragma unroll
  for (int g = 0; g < NHEAD; ++g) {
    rm[g] = rowmax[(size_t)(b * NHEAD + g) * NSEQ + n0 + tn];
    const float s = rowsum[(size_t)(b * NHEAD + g) * NSEQ + n0 + tn];
    rsi[g] = (s > 0.f) ? (1.0f / s) : 0.f;
  }

  const __bf16* qrow = q + ((size_t)(b * NHEAD + h) * NSEQ + n0 + l16) * HD;
  const v16bf qf0 = ld_frag(qrow, hlf, 0);
  const v16bf qf1 = ld_frag(qrow, hlf, 32);
  __syncthreads();

  v8f oacc[4];
#pragma unroll
  for (int jd = 0; jd < 4; ++jd) oacc[jd] = zero_v8f();

  for (int m0 = 0; m0 < NSEQ; m0 += 64) {
    // -- recompute scores (wave h)
    v8f sacc[4];
#pragma unroll
    for (int j = 0; j < 4; ++j) sacc[j] = zero_v8f();
#pragma unroll
    for (int j = 0; j < 4; ++j) {
      const __bf16* krow =
          kmat + ((size_t)(b * NHEAD + h) * NSEQ + m0 + j * 16 + l16) * HD;
      sacc[j] = wmma_bf16(qf0, ld_frag(krow, hlf, 0), sacc[j]);
      sacc[j] = wmma_bf16(qf1, ld_frag(krow, hlf, 32), sacc[j]);
    }
#pragma unroll
    for (int j = 0; j < 4; ++j)
#pragma unroll
      for (int r = 0; r < 8; ++r)
        s_lds[h][r + 8 * hlf][j * 16 + l16] = sacc[j][r];
    __syncthreads();

    // -- mix1 + interaction + mask + softmax -> p, write pure_attention
    float p[NHEAD][4];
#pragma unroll
    for (int mi = 0; mi < 4; ++mi) {
      const int m = m0 + tm + mi;
      float sh[NHEAD];
#pragma unroll
      for (int hh = 0; hh < NHEAD; ++hh) sh[hh] = s_lds[hh][tn][tm + mi];
      const float* ip =
          interaction + (((size_t)(b * NSEQ + n0 + tn)) * NSEQ + m) * NHEAD;
      const float4 i0 = ((const float4*)ip)[0];
      const float4 i1 = ((const float4*)ip)[1];
      const float iv[NHEAD] = {i0.x, i0.y, i0.z, i0.w, i1.x, i1.y, i1.z, i1.w};
      const unsigned char mk = mask[((size_t)b * NSEQ + n0 + tn) * NSEQ + m];
#pragma unroll
      for (int g = 0; g < NHEAD; ++g) {
        float v = bt1s[g];
#pragma unroll
        for (int hh = 0; hh < NHEAD; ++hh) v += sh[hh] * wt1[hh][g];
        v += iv[g];
        const float lg = mk ? v : -1.0e9f;
        p[g][mi] = __expf(lg - rm[g]) * rsi[g];
      }
    }
#pragma unroll
    for (int g = 0; g < NHEAD; ++g) {
      float4 o;
      o.x = p[g][0]; o.y = p[g][1]; o.z = p[g][2]; o.w = p[g][3];
      *(float4*)(pure_attn +
                 ((size_t)(b * NHEAD + g) * NSEQ + n0 + tn) * NSEQ + m0 + tm) = o;
    }
    // -- mix2 -> bf16 attn fragments in LDS
#pragma unroll
    for (int mi = 0; mi < 4; ++mi) {
#pragma unroll
      for (int g = 0; g < NHEAD; ++g) {
        float v = bt2s[g];
#pragma unroll
        for (int hh = 0; hh < NHEAD; ++hh) v += p[hh][mi] * wt2[hh][g];
        a_lds[g][tn][tm + mi] = (__bf16)v;
      }
    }
    __syncthreads();

    // -- attn2_h(16 x 64) @ V_h(64 x 64): wave h, vT rows contiguous in m
#pragma unroll
    for (int ks = 0; ks < 2; ++ks) {
      const v16bf af = ld_frag(&a_lds[h][l16][0], hlf, ks * 32);
#pragma unroll
      for (int jd = 0; jd < 4; ++jd) {
        const __bf16* vrow =
            vt + ((size_t)(b * NHEAD + h) * HD + jd * 16 + l16) * NSEQ + m0;
        oacc[jd] = wmma_bf16(af, ld_frag(vrow, hlf, ks * 32), oacc[jd]);
      }
    }
    __syncthreads();
  }

  // -- attended[b][n][h*64 + d]
#pragma unroll
  for (int jd = 0; jd < 4; ++jd) {
    const int d = jd * 16 + l16;
#pragma unroll
    for (int r = 0; r < 8; ++r) {
      const int n = n0 + r + 8 * hlf;
      attended[((size_t)b * NSEQ + n) * DIMC + h * HD + d] = oacc[jd][r];
    }
  }
}

// ---------------------------------------------------------------------------
extern "C" void kernel_launch(void* const* d_in, const int* in_sizes, int n_in,
                              void* d_out, int out_size, void* d_ws, size_t ws_size,
                              hipStream_t stream) {
  const float* inputs       = (const float*)d_in[0];
  const unsigned char* mask = (const unsigned char*)d_in[1];
  const float* interaction  = (const float*)d_in[2];
  const float* W_qkv        = (const float*)d_in[3];
  const float* b_qkv        = (const float*)d_in[4];
  const float* W_t1         = (const float*)d_in[5];
  const float* b_t1         = (const float*)d_in[6];
  const float* W_t2         = (const float*)d_in[7];
  const float* b_t2         = (const float*)d_in[8];
  const float* W_out        = (const float*)d_in[9];
  const float* b_out        = (const float*)d_in[10];
  (void)in_sizes; (void)n_in; (void)out_size;

  float* out       = (float*)d_out;                          // (B,N,DIM)
  float* pure_attn = out + (size_t)BQ * NSEQ * DIMC;         // (B,H,N,N)

  char* ws = (char*)d_ws;
  size_t off = 0;
  auto wsalloc = [&](size_t bytes) -> void* {
    off = (off + 255) & ~(size_t)255;
    void* p = ws + off;
    off += bytes;
    return p;
  };
  const size_t qkv_elems = (size_t)BQ * NHEAD * NSEQ * HD;   // 8,388,608
  __bf16* qws  = (__bf16*)wsalloc(qkv_elems * sizeof(__bf16));
  __bf16* kws  = (__bf16*)wsalloc(qkv_elems * sizeof(__bf16));
  __bf16* vtws = (__bf16*)wsalloc(qkv_elems * sizeof(__bf16));
  float* rowmax   = (float*)wsalloc((size_t)BQ * NHEAD * NSEQ * sizeof(float));
  float* rowsum   = (float*)wsalloc((size_t)BQ * NHEAD * NSEQ * sizeof(float));
  float* attended = (float*)wsalloc((size_t)BQ * NSEQ * DIMC * sizeof(float));
  (void)ws_size;

  const dim3 blk(256);
  // 1) QKV projection: (16384 x 512) @ (512 x 1536)
  gemm_bf16_kernel<3 * DIMC, 0><<<dim3((3 * DIMC) / 64, (BQ * NSEQ) / 128), blk, 0,
                                  stream>>>(inputs, W_qkv, b_qkv, qws, kws, vtws,
                                            nullptr);
  // 2) softmax row statistics
  attn_stats_kernel<<<dim3(NSEQ / 16, BQ), blk, 0, stream>>>(
      qws, kws, interaction, mask, W_t1, b_t1, rowmax, rowsum);
  // 3) softmax + pure_attention + mix2 + attn@V
  attn_apply_kernel<<<dim3(NSEQ / 16, BQ), blk, 0, stream>>>(
      qws, kws, vtws, interaction, mask, W_t1, b_t1, W_t2, b_t2, rowmax, rowsum,
      pure_attn, attended);
  // 4) output projection: (16384 x 512) @ (512 x 512)
  gemm_bf16_kernel<DIMC, 1><<<dim3(DIMC / 64, (BQ * NSEQ) / 128), blk, 0, stream>>>(
      attended, W_out, b_out, nullptr, nullptr, nullptr, out);
}
